// GeneralConv_7533372637985
// MI455X (gfx1250) — compile-verified, matrix operand order checked by reading, not run on hardware
//
#include <hip/hip_runtime.h>
#include <math.h>

// Problem constants (match reference)
#define T_   3
#define R_   5
#define H_   8
#define IN_  256
#define OUT_ 256
#define DK_  32
#define N_   30000
#define E_   200000
#define ML_  240

typedef __attribute__((ext_vector_type(16))) _Float16 v16h;
typedef __attribute__((ext_vector_type(8)))  float    v8f;

// K index inside a 16-bit WMMA A/B fragment (CDNA5 ISA 7.12.2):
// VGPR v (= s>>1) in 0..3 holds K = half*8 + 2v + (s&1); VGPRs 4..7 add 16.
__device__ __forceinline__ int kmap(int s, int half) {
  int v = s >> 1;
  return ((v >> 2) << 4) | (half << 3) | ((v & 3) << 1) | (s & 1);
}

__device__ __forceinline__ void atomic_max_f32(float* addr, float val) {
  if (val >= 0.0f) atomicMax((int*)addr, __float_as_int(val));
  else             atomicMin((unsigned int*)addr, __float_as_uint(val));
}

// ---------------------------------------------------------------------------
// Pack a [T][256][256] f32 weight matrix into WMMA B-fragment order (f16):
// Wp[((t*16+ct)*8+kk)*512 + lane*16 + s] = W[t][kk*32 + kmap(s,lane>>4)][ct*16 + (lane&15)]
// so each B fragment is one contiguous 32B v16h per lane.
// ---------------------------------------------------------------------------
__global__ void pack_w_kernel(const float* __restrict__ W, _Float16* __restrict__ Wp,
                              int total) {
  int idx = blockIdx.x * 256 + threadIdx.x;
  if (idx >= total) return;
  int s    = idx & 15;
  int lane = (idx >> 4) & 31;
  int kk   = (idx >> 9) & 7;
  int ct   = (idx >> 12) & 15;
  int t    = idx >> 16;
  int half = lane >> 4;
  int n    = lane & 15;
  int k    = kk * 32 + kmap(s, half);
  Wp[idx] = (_Float16)W[(size_t)t * (IN_ * OUT_) + (size_t)k * OUT_ + ct * 16 + n];
}

// ---------------------------------------------------------------------------
__global__ void init_kernel(float* __restrict__ nmax, float* __restrict__ nden,
                            float* __restrict__ aggr) {
  int i = blockIdx.x * blockDim.x + threadIdx.x;
  int stride = gridDim.x * blockDim.x;
  for (int k = i; k < N_ * H_; k += stride) { nmax[k] = -__builtin_inff(); nden[k] = 0.0f; }
  for (size_t k = i; k < (size_t)N_ * OUT_; k += (size_t)stride) aggr[k] = 0.0f;
}

// ---------------------------------------------------------------------------
// rte_vec[m][i] = dot(rte_emb[m], rte_W[i]) + rte_b[i]     ([240,256])
// ---------------------------------------------------------------------------
__global__ void rte_vec_kernel(const float* __restrict__ emb, const float* __restrict__ Wt,
                               const float* __restrict__ b, float* __restrict__ out) {
  int idx = blockIdx.x * 256 + threadIdx.x;     // < ML_*IN_
  int i = idx & 255, m = idx >> 8;
  float acc = b[i];
  const float* e = emb + (size_t)m * IN_;
  const float* w = Wt + (size_t)i * IN_;
  for (int j = 0; j < IN_; ++j) acc = fmaf(e[j], w[j], acc);
  out[idx] = acc;
}

// rte_k[t][m][o] = dot(rte_vec[m], Wk[t][:,o]);  same for rte_v with Wv.
__global__ void rte_kv_kernel(const float* __restrict__ rv,
                              const float* __restrict__ Wk, const float* __restrict__ Wv,
                              float* __restrict__ rk, float* __restrict__ rvv) {
  int idx = blockIdx.x * 256 + threadIdx.x;     // < T_*ML_*OUT_
  int o = idx & 255;
  int m = (idx >> 8) % ML_;
  int t = idx / (ML_ * OUT_);
  const float* v  = rv + (size_t)m * IN_;
  const float* wk = Wk + (size_t)t * (IN_ * OUT_);
  const float* wv = Wv + (size_t)t * (IN_ * OUT_);
  float ak = 0.0f, av = 0.0f;
  for (int i = 0; i < IN_; ++i) {
    float x = v[i];
    ak = fmaf(x, wk[(size_t)i * OUT_ + o], ak);
    av = fmaf(x, wv[(size_t)i * OUT_ + o], av);
  }
  rk[idx] = ak;
  rvv[idx] = av;
}

// ---------------------------------------------------------------------------
// One 16-row tile GEMM: Out[16,256] = A(16,256) * W[t](256,256) + bias[t],
// rows written only where node type == t.  Asv = A tile in fragment order,
// Wpk = packed weights, both one v16h per (k-step, lane).
// ---------------------------------------------------------------------------
__device__ __forceinline__ void project_tile(
    const v16h* __restrict__ Asv, const v16h* __restrict__ Wpk,
    const float* __restrict__ bias, float* __restrict__ Out,
    const int* __restrict__ stype, int tmask, int base, int wave, int lane) {
  const int n = lane & 15, hs = lane >> 4;
  for (int t = 0; t < T_; ++t) {
    if (!((tmask >> t) & 1)) continue;
#pragma unroll
    for (int c = 0; c < 2; ++c) {
      const int ct = wave * 2 + c;
      const int nbase = ct * 16;
      const float bv = bias[t * OUT_ + nbase + n];
      v8f acc;
#pragma unroll
      for (int i = 0; i < 8; ++i) acc[i] = bv;
#pragma unroll
      for (int kk = 0; kk < 8; ++kk) {
        v16h a = Asv[kk * 32 + lane];
        v16h b = Wpk[(((size_t)t * 16 + ct) * 8 + kk) * 32 + lane];
        acc = __builtin_amdgcn_wmma_f32_16x16x32_f16(false, a, false, b,
                                                     (short)0, acc, false, false);
      }
#pragma unroll
      for (int i = 0; i < 8; ++i) {
        int m = i + hs * 8;
        if (stype[m] == t)
          Out[(size_t)(base + m) * OUT_ + nbase + n] = acc[i];
      }
    }
  }
}

// Q/K/V projection: one block = 16 nodes, 8 waves x 2 column tiles = 256 cols.
__global__ __launch_bounds__(256) void qkv_wmma_kernel(
    const float* __restrict__ x,
    const _Float16* __restrict__ wq16, const _Float16* __restrict__ wk16,
    const _Float16* __restrict__ wv16,
    const float* __restrict__ bq, const float* __restrict__ bk, const float* __restrict__ bv,
    const int* __restrict__ ntype,
    float* __restrict__ qn, float* __restrict__ kn, float* __restrict__ vn) {
  __shared__ __align__(32) _Float16 As[16 * IN_];
  __shared__ int stype[16];
  const int tid = threadIdx.x;
  const int base = blockIdx.x * 16;
  // Pack the 16xIN A tile into fragment order (f32 -> f16).
#pragma unroll
  for (int j = 0; j < 16; ++j) {
    int flat = j * 256 + tid;                 // = (kk*32+lane)*16 + s
    int s = flat & 15, ln = (flat >> 4) & 31, kk = flat >> 9;
    int k = kk * 32 + kmap(s, ln >> 4);
    As[flat] = (_Float16)x[(size_t)(base + (ln & 15)) * IN_ + k];
  }
  if (tid < 16) stype[tid] = ntype[base + tid];
  __syncthreads();
  int tmask = 0;
#pragma unroll
  for (int i = 0; i < 16; ++i) tmask |= 1 << stype[i];
  const int wave = tid >> 5, lane = tid & 31;
  const v16h* Asv = (const v16h*)As;
  project_tile(Asv, (const v16h*)wq16, bq, qn, stype, tmask, base, wave, lane);
  project_tile(Asv, (const v16h*)wk16, bk, kn, stype, tmask, base, wave, lane);
  project_tile(Asv, (const v16h*)wv16, bv, vn, stype, tmask, base, wave, lane);
}

// ---------------------------------------------------------------------------
// Edge phase 1: one wave per edge.  k_e = kn[src] + rte_k[st,time];
// per head: k_t[f] = sum_d k_e[h*32+d]*rel_att[r,h,d,f];  logit = <q_e, k_t>.
// ---------------------------------------------------------------------------
__global__ __launch_bounds__(256) void edge_logit_kernel(
    const float* __restrict__ qn, const float* __restrict__ kn,
    const float* __restrict__ rteK,
    const int* __restrict__ ei, const int* __restrict__ etype,
    const int* __restrict__ etime, const int* __restrict__ ntype,
    const float* __restrict__ rel_att, const float* __restrict__ rel_pri,
    float* __restrict__ logit, float* __restrict__ nmax) {
  __shared__ float sk[8][IN_];
  const int lane = threadIdx.x & 31, we = threadIdx.x >> 5;
  const int e = blockIdx.x * 8 + we;          // grid is exactly E_/8
  const int src = ei[e], dst = ei[E_ + e];
  const int r = etype[e], tm = etime[e], st = ntype[src];
  const float* kb = kn + (size_t)src * OUT_;
  const float* rb = rteK + ((size_t)st * ML_ + tm) * OUT_;
#pragma unroll
  for (int j = 0; j < 8; ++j) {
    int idx = j * 32 + lane;
    sk[we][idx] = kb[idx] + rb[idx];
  }
  __syncthreads();
  const float* qb = qn + (size_t)dst * OUT_;
  const float* A = rel_att + (size_t)r * H_ * DK_ * DK_;
  for (int h = 0; h < H_; ++h) {
    float acc = 0.0f;
    const float* Ah = A + h * DK_ * DK_;
#pragma unroll
    for (int d = 0; d < DK_; ++d)
      acc = fmaf(sk[we][h * DK_ + d], Ah[d * DK_ + lane], acc);
    float p = acc * qb[h * DK_ + lane];
#pragma unroll
    for (int off = 16; off > 0; off >>= 1) p += __shfl_xor(p, off, 32);
    if (lane == 0) {
      float lg = p * rel_pri[r * H_ + h] * 0.17677669529663689f; // 1/sqrt(32)
      logit[(size_t)e * H_ + h] = lg;
      atomic_max_f32(&nmax[(size_t)dst * H_ + h], lg);
    }
  }
}

// Edge phase 2: ex = exp(logit - max[dst]); accumulate denominator.
__global__ void edge_den_kernel(float* __restrict__ logit, const int* __restrict__ ei,
                                const float* __restrict__ nmax, float* __restrict__ nden) {
  int idx = blockIdx.x * 256 + threadIdx.x;   // grid exactly E_*H_/256
  int e = idx >> 3, h = idx & 7;
  int dst = ei[E_ + e];
  float ex = __expf(logit[idx] - nmax[(size_t)dst * H_ + h]);
  logit[idx] = ex;                            // reuse buffer for ex
  atomicAdd(&nden[(size_t)dst * H_ + h], ex);
}

// Edge phase 3: v_t = (vn[src]+rte_v) @ rel_msg[r];  aggr[dst] += v_t * att.
__global__ __launch_bounds__(256) void edge_aggr_kernel(
    const float* __restrict__ vn, const float* __restrict__ rteV,
    const int* __restrict__ ei, const int* __restrict__ etype,
    const int* __restrict__ etime, const int* __restrict__ ntype,
    const float* __restrict__ rel_msg, const float* __restrict__ exbuf,
    const float* __restrict__ nden, float* __restrict__ aggr) {
  __shared__ float sv[8][IN_];
  const int lane = threadIdx.x & 31, we = threadIdx.x >> 5;
  const int e = blockIdx.x * 8 + we;
  const int src = ei[e], dst = ei[E_ + e];
  const int r = etype[e], tm = etime[e], st = ntype[src];
  const float* vb = vn + (size_t)src * OUT_;
  const float* rb = rteV + ((size_t)st * ML_ + tm) * OUT_;
#pragma unroll
  for (int j = 0; j < 8; ++j) {
    int idx = j * 32 + lane;
    sv[we][idx] = vb[idx] + rb[idx];
  }
  __syncthreads();
  const float* M = rel_msg + (size_t)r * H_ * DK_ * DK_;
  for (int h = 0; h < H_; ++h) {
    float acc = 0.0f;
    const float* Mh = M + h * DK_ * DK_;
#pragma unroll
    for (int d = 0; d < DK_; ++d)
      acc = fmaf(sv[we][h * DK_ + d], Mh[d * DK_ + lane], acc);
    float att = exbuf[(size_t)e * H_ + h] / nden[(size_t)dst * H_ + h];
    atomicAdd(&aggr[(size_t)dst * OUT_ + h * DK_ + lane], acc * att);
  }
}

// ---------------------------------------------------------------------------
// Final: trans = aggr @ Wa[type] + ba;  mixed = trans*alpha + x*(1-alpha);
// out = layernorm(mixed)*ln_g[type] + ln_b[type].
// ---------------------------------------------------------------------------
__global__ __launch_bounds__(256) void out_wmma_kernel(
    const float* __restrict__ aggr, const _Float16* __restrict__ wa16,
    const float* __restrict__ ba, const float* __restrict__ x,
    const int* __restrict__ ntype, const float* __restrict__ skip,
    const float* __restrict__ ln_g, const float* __restrict__ ln_b,
    float* __restrict__ out) {
  __shared__ __align__(32) _Float16 As[16 * OUT_];
  __shared__ float Tr[16 * OUT_];
  __shared__ int stype[16];
  __shared__ float alpha_s[16], mu_s[16], rs_s[16];
  const int tid = threadIdx.x;
  const int base = blockIdx.x * 16;
#pragma unroll
  for (int j = 0; j < 16; ++j) {
    int flat = j * 256 + tid;
    int s = flat & 15, ln = (flat >> 4) & 31, kk = flat >> 9;
    int k = kk * 32 + kmap(s, ln >> 4);
    As[flat] = (_Float16)aggr[(size_t)(base + (ln & 15)) * OUT_ + k];
  }
  if (tid < 16) {
    int t = ntype[base + tid];
    stype[tid] = t;
    alpha_s[tid] = 1.0f / (1.0f + __expf(-skip[t]));
  }
  __syncthreads();
  int tmask = 0;
#pragma unroll
  for (int i = 0; i < 16; ++i) tmask |= 1 << stype[i];
  const int wave = tid >> 5, lane = tid & 31;
  const int n = lane & 15, hs = lane >> 4;
  const v16h* Asv = (const v16h*)As;
  const v16h* Wpk = (const v16h*)wa16;
  for (int t = 0; t < T_; ++t) {
    if (!((tmask >> t) & 1)) continue;
#pragma unroll
    for (int c = 0; c < 2; ++c) {
      const int ct = wave * 2 + c;
      const int nbase = ct * 16;
      const float bv = ba[t * OUT_ + nbase + n];
      v8f acc;
#pragma unroll
      for (int i = 0; i < 8; ++i) acc[i] = bv;
#pragma unroll
      for (int kk = 0; kk < 8; ++kk) {
        v16h a = Asv[kk * 32 + lane];
        v16h b = Wpk[(((size_t)t * 16 + ct) * 8 + kk) * 32 + lane];
        acc = __builtin_amdgcn_wmma_f32_16x16x32_f16(false, a, false, b,
                                                     (short)0, acc, false, false);
      }
#pragma unroll
      for (int i = 0; i < 8; ++i) {
        int m = i + hs * 8;
        if (stype[m] == t) Tr[m * OUT_ + nbase + n] = acc[i];
      }
    }
  }
  __syncthreads();
  // mixed = trans*alpha + x*(1-alpha), in place in Tr
#pragma unroll
  for (int j = 0; j < 16; ++j) {
    int flat = j * 256 + tid;
    int row = flat >> 8, col = flat & 255;
    float a = alpha_s[row];
    Tr[flat] = Tr[flat] * a + x[(size_t)(base + row) * IN_ + col] * (1.0f - a);
  }
  __syncthreads();
  if (tid < 16) {
    float mu = 0.0f;
    for (int c = 0; c < OUT_; ++c) mu += Tr[tid * OUT_ + c];
    mu *= (1.0f / OUT_);
    float var = 0.0f;
    for (int c = 0; c < OUT_; ++c) {
      float d = Tr[tid * OUT_ + c] - mu;
      var = fmaf(d, d, var);
    }
    var *= (1.0f / OUT_);
    mu_s[tid] = mu;
    rs_s[tid] = rsqrtf(var + 1e-5f);
  }
  __syncthreads();
#pragma unroll
  for (int j = 0; j < 16; ++j) {
    int flat = j * 256 + tid;
    int row = flat >> 8, col = flat & 255;
    int t = stype[row];
    out[(size_t)(base + row) * OUT_ + col] =
        (Tr[flat] - mu_s[row]) * rs_s[row] * ln_g[t * OUT_ + col] + ln_b[t * OUT_ + col];
  }
}

// ---------------------------------------------------------------------------
extern "C" void kernel_launch(void* const* d_in, const int* in_sizes, int n_in,
                              void* d_out, int out_size, void* d_ws, size_t ws_size,
                              hipStream_t stream) {
  (void)in_sizes; (void)n_in; (void)out_size; (void)ws_size;
  const float* node_inp  = (const float*)d_in[0];
  const float* Wk        = (const float*)d_in[1];
  const float* bk        = (const float*)d_in[2];
  const float* Wq        = (const float*)d_in[3];
  const float* bq        = (const float*)d_in[4];
  const float* Wv        = (const float*)d_in[5];
  const float* bv        = (const float*)d_in[6];
  const float* Wa        = (const float*)d_in[7];
  const float* ba        = (const float*)d_in[8];
  const float* ln_g      = (const float*)d_in[9];
  const float* ln_b      = (const float*)d_in[10];
  const float* rel_pri   = (const float*)d_in[11];
  const float* rel_att   = (const float*)d_in[12];
  const float* rel_msg   = (const float*)d_in[13];
  const float* skip      = (const float*)d_in[14];
  const float* rte_W     = (const float*)d_in[15];
  const float* rte_b     = (const float*)d_in[16];
  const float* rte_emb   = (const float*)d_in[17];
  const int*   node_type = (const int*)d_in[18];
  const int*   edge_idx  = (const int*)d_in[19];
  const int*   edge_type = (const int*)d_in[20];
  const int*   edge_time = (const int*)d_in[21];
  float* outp = (float*)d_out;

  // Workspace bump allocator (256B aligned slices).
  char* p = (char*)d_ws;
  auto alloc = [&](size_t bytes) -> void* {
    void* r = (void*)p;
    p += (bytes + 255) & ~(size_t)255;
    return r;
  };
  float* rte_vec = (float*)alloc((size_t)ML_ * IN_ * 4);
  float* rteK    = (float*)alloc((size_t)T_ * ML_ * OUT_ * 4);
  float* rteV    = (float*)alloc((size_t)T_ * ML_ * OUT_ * 4);
  float* qn      = (float*)alloc((size_t)N_ * OUT_ * 4);
  float* kn      = (float*)alloc((size_t)N_ * OUT_ * 4);
  float* vn      = (float*)alloc((size_t)N_ * OUT_ * 4);
  float* logit   = (float*)alloc((size_t)E_ * H_ * 4);
  float* nmax    = (float*)alloc((size_t)N_ * H_ * 4);
  float* nden    = (float*)alloc((size_t)N_ * H_ * 4);
  float* aggr    = (float*)alloc((size_t)N_ * OUT_ * 4);
  _Float16* wq16 = (_Float16*)alloc((size_t)T_ * IN_ * OUT_ * 2);
  _Float16* wk16 = (_Float16*)alloc((size_t)T_ * IN_ * OUT_ * 2);
  _Float16* wv16 = (_Float16*)alloc((size_t)T_ * IN_ * OUT_ * 2);
  _Float16* wa16 = (_Float16*)alloc((size_t)T_ * OUT_ * OUT_ * 2);

  const int wtotal = T_ * IN_ * OUT_;           // 196608
  pack_w_kernel<<<wtotal / 256, 256, 0, stream>>>(Wq, wq16, wtotal);
  pack_w_kernel<<<wtotal / 256, 256, 0, stream>>>(Wk, wk16, wtotal);
  pack_w_kernel<<<wtotal / 256, 256, 0, stream>>>(Wv, wv16, wtotal);
  pack_w_kernel<<<wtotal / 256, 256, 0, stream>>>(Wa, wa16, wtotal);

  init_kernel<<<4096, 256, 0, stream>>>(nmax, nden, aggr);

  rte_vec_kernel<<<(ML_ * IN_) / 256, 256, 0, stream>>>(rte_emb, rte_W, rte_b, rte_vec);
  rte_kv_kernel<<<(T_ * ML_ * OUT_) / 256, 256, 0, stream>>>(rte_vec, Wk, Wv, rteK, rteV);

  qkv_wmma_kernel<<<N_ / 16, 256, 0, stream>>>(node_inp, wq16, wk16, wv16,
                                               bq, bk, bv, node_type, qn, kn, vn);

  edge_logit_kernel<<<E_ / 8, 256, 0, stream>>>(qn, kn, rteK, edge_idx, edge_type,
                                                edge_time, node_type, rel_att, rel_pri,
                                                logit, nmax);
  edge_den_kernel<<<(E_ * H_) / 256, 256, 0, stream>>>(logit, edge_idx, nmax, nden);
  edge_aggr_kernel<<<E_ / 8, 256, 0, stream>>>(vn, rteV, edge_idx, edge_type, edge_time,
                                               node_type, rel_msg, logit, nden, aggr);

  out_wmma_kernel<<<N_ / 16, 256, 0, stream>>>(aggr, wa16, ba, node_inp, node_type,
                                               skip, ln_g, ln_b, outp);
}
